// MemUnit_7868380086597
// MI455X (gfx1250) — matrix operand order repaired and checked by reading, not compiled, not run on hardware
//
#include <hip/hip_runtime.h>

#define BATCH 8192
#define DIM   256
#define KCOLS 65536

#define M_TILE 256          // 8 waves x 2 slabs x 16 rows
#define N_TILE 256          // columns per LDS tile
#define CTILES 8            // column tiles processed per block (double-buffered)
#define WSTRIDE 528         // bytes per n-row in LDS (512B bf16 + 16B pad), multiple of 16
#define LDSBUF  (N_TILE * WSTRIDE)   // 135168 B per buffer

typedef __attribute__((ext_vector_type(16))) __bf16      v16bf;
typedef __attribute__((ext_vector_type(8)))  float        v8f;
typedef __attribute__((ext_vector_type(4)))  unsigned int v4u;
typedef __attribute__((ext_vector_type(8)))  int          v8i;
typedef __attribute__((ext_vector_type(4)))  int          v4i;

#if defined(__has_builtin)
#if __has_builtin(__builtin_amdgcn_tensor_load_to_lds) && __has_builtin(__builtin_amdgcn_s_wait_tensorcnt)
#define USE_TDM 1
#endif
#endif
#ifndef USE_TDM
#define USE_TDM 0
#endif

static __device__ __forceinline__ unsigned short f2bf(float f) {
    unsigned int u = __float_as_uint(f);
    u += 0x7FFFu + ((u >> 16) & 1u);
    return (unsigned short)(u >> 16);
}

// ---------- pre-pass 1: Wt[k][d] = bf16(W[d][k]); cninv[k] = 1/||W[:,k]||; init best ----------
__global__ void prepW_k(const float* __restrict__ W, unsigned short* __restrict__ Wt,
                        float* __restrict__ cninv, unsigned long long* __restrict__ best) {
    int k = blockIdx.x * 256 + threadIdx.x;
    float s = 0.f;
    for (int d8 = 0; d8 < DIM / 8; ++d8) {
        union { unsigned short h[8]; v4u u; } pk;
#pragma unroll
        for (int i = 0; i < 8; ++i) {
            float w = W[(size_t)(d8 * 8 + i) * KCOLS + k];
            s += w * w;
            pk.h[i] = f2bf(w);
        }
        *(v4u*)(Wt + (size_t)k * DIM + d8 * 8) = pk.u;
    }
    cninv[k] = 1.0f / sqrtf(s);
    if (k < BATCH) best[k] = 0ull;
}

// ---------- pre-pass 2: Tb = bf16(t) ----------
__global__ void prepT_k(const float* __restrict__ T, unsigned short* __restrict__ Tb) {
    int i = blockIdx.x * 256 + threadIdx.x;
    const float* p = T + (size_t)i * 8;
    union { unsigned short h[8]; v4u u; } pk;
#pragma unroll
    for (int j = 0; j < 8; ++j) pk.h[j] = f2bf(p[j]);
    *(v4u*)(Tb + (size_t)i * 8) = pk.u;
}

#if USE_TDM
// One TDM descriptor: load a 256-row x 512B tile of Wt into LDS, inserting 16B of
// padding after every 512B row (reproduces WSTRIDE=528 exactly).
static __device__ __forceinline__ void tdm_load_tile(unsigned lds_off,
                                                     const unsigned short* gsrc) {
    unsigned long long ga = (unsigned long long)(size_t)gsrc;
    v4u g0;
    g0.x = 1u;                                            // count=1, user mode
    g0.y = lds_off;                                       // lds_addr (bytes)
    g0.z = (unsigned)(ga & 0xffffffffu);                  // global_addr[31:0]
    g0.w = (unsigned)((ga >> 32) & 0x01ffffffu) | 0x80000000u; // addr[56:32] | type=2
    v8i g1;
    g1[0] = (3 << 16)            // data_size = 8 bytes
          | (1 << 20)            // pad_enable
          | (6 << 22)            // pad_interval: 128 DWORDs (512B)
          | (3 << 25);           // pad_amount: 4 DWORDs (16B)
    g1[1] = (64 << 16);          // tensor_dim0 = 64 (8B units; 512B row)
    g1[2] = 0;                   // tensor_dim0 hi, tensor_dim1 lo16 (65536 & 0xffff = 0)
    g1[3] = 1 | (64 << 16);      // tensor_dim1 hi16 = 1 (=> 65536), tile_dim0 = 64
    g1[4] = 256;                 // tile_dim1 = 256 rows (tile_dim2 = 0)
    g1[5] = 64;                  // tensor_dim0_stride = 64 (8B units)
    g1[6] = (64 << 16);          // dim0_stride hi = 0, tensor_dim1_stride lo16 = 64
    g1[7] = 0;
    v4i z4 = (v4i)(0);
    v8i z8 = (v8i)(0);
    // clang-23 (therock) 6-arg form: (g0, g1, g2, g3, g4[unused/VADDR4], cpol)
    __builtin_amdgcn_tensor_load_to_lds(g0, g1, z4, z4, z8, 0);
}
#endif

// ---------- main: bf16 WMMA GEMM + scaled argmax, TDM double-buffered ----------
__global__ __launch_bounds__(256, 1)
void simargmax_k(const unsigned short* __restrict__ Tb,
                 const unsigned short* __restrict__ Wt,
                 const float* __restrict__ cninv,
                 unsigned long long* __restrict__ best) {
    __shared__ __align__(16) unsigned char ldsW[2][LDSBUF];   // 264 KB

    const int tid = threadIdx.x;
    const int n0  = blockIdx.x * (N_TILE * CTILES);
    const int m0  = blockIdx.y * M_TILE;

    const int wave = tid >> 5;
    const int lane = tid & 31;
    const int half = lane >> 4;
    const int l15  = lane & 15;
    const int sel8 = half * 8;

#if USE_TDM
    if (wave == 0) {
        unsigned lds0 = (unsigned)(size_t)&ldsW[0][0];
        tdm_load_tile(lds0,          Wt + (size_t)n0 * DIM);
        tdm_load_tile(lds0 + LDSBUF, Wt + (size_t)(n0 + N_TILE) * DIM);
    }
#endif

    // ---- 2 resident A slabs per wave (held across all 8 column tiles) ----
    v16bf afrag[2][8];
#pragma unroll
    for (int s = 0; s < 2; ++s) {
        const int rowA = m0 + wave * 32 + s * 16 + l15;
#pragma unroll
        for (int kc = 0; kc < 8; ++kc) {
            const unsigned short* pa = Tb + (size_t)rowA * DIM + kc * 32 + sel8;
            union { v4u u2[2]; v16bf v; } a;
            a.u2[0] = *(const v4u*)(pa);
            a.u2[1] = *(const v4u*)(pa + 16);
            afrag[s][kc] = a.v;
        }
    }

    float        bestv[2][8];
    unsigned int besti[2][8];
#pragma unroll
    for (int s = 0; s < 2; ++s)
#pragma unroll
        for (int v = 0; v < 8; ++v) { bestv[s][v] = -3.0e38f; besti[s][v] = 0u; }

    for (int ct = 0; ct < CTILES; ++ct) {
#if USE_TDM
        if (wave == 0) {
            if (ct < CTILES - 1) __builtin_amdgcn_s_wait_tensorcnt(1); // tile ct landed
            else                 __builtin_amdgcn_s_wait_tensorcnt(0);
        }
        __syncthreads();                       // tile ct visible to all waves
#else
        {   // synchronous staging fallback
            const unsigned short* src = Wt + (size_t)(n0 + ct * N_TILE + tid) * DIM;
            unsigned char* dst = &ldsW[ct & 1][(size_t)tid * WSTRIDE];
#pragma unroll 4
            for (int i = 0; i < 32; ++i)
                *(v4u*)(dst + i * 16) = *(const v4u*)(src + i * 8);
            __syncthreads();
        }
#endif
        const unsigned char* lbase = &ldsW[ct & 1][0];
        const int colbase = n0 + ct * N_TILE;

        for (int j = 0; j < 16; ++j) {
            v8f acc0 = {}, acc1 = {};
#pragma unroll
            for (int kc = 0; kc < 8; ++kc) {
                const unsigned char* pb =
                    lbase + (size_t)(j * 16 + l15) * WSTRIDE + (kc * 32 + half * 8) * 2;
                union { v4u u2[2]; v16bf v; } b;
                b.u2[0] = *(const v4u*)(pb);
                b.u2[1] = *(const v4u*)(pb + 32);
                acc0 = __builtin_amdgcn_wmma_f32_16x16x32_bf16(
                    false, afrag[0][kc], false, b.v, (short)0, acc0, false, false);
                acc1 = __builtin_amdgcn_wmma_f32_16x16x32_bf16(
                    false, afrag[1][kc], false, b.v, (short)0, acc1, false, false);
            }
            const int   col = colbase + j * 16 + l15;
            const float ci  = cninv[col];
#pragma unroll
            for (int v = 0; v < 8; ++v) {
                float v0 = acc0[v] * ci;
                float v1 = acc1[v] * ci;
                if (v0 > bestv[0][v]) { bestv[0][v] = v0; besti[0][v] = (unsigned)col; }
                if (v1 > bestv[1][v]) { bestv[1][v] = v1; besti[1][v] = (unsigned)col; }
            }
        }

        __syncthreads();                       // all waves done reading buf[ct&1]
#if USE_TDM
        if (wave == 0 && ct + 2 < CTILES) {
            unsigned lds = (unsigned)(size_t)&ldsW[ct & 1][0];
            tdm_load_tile(lds, Wt + (size_t)(n0 + (ct + 2) * N_TILE) * DIM);
        }
#endif
    }

    // ---- argmax across the 16 lanes sharing each row, then global merge ----
#pragma unroll
    for (int s = 0; s < 2; ++s)
#pragma unroll
    for (int v = 0; v < 8; ++v) {
        float        bv = bestv[s][v];
        unsigned int bi = besti[s][v];
#pragma unroll
        for (int m = 1; m < 16; m <<= 1) {
            float        ov = __shfl_xor(bv, m, 32);
            unsigned int oi = (unsigned int)__shfl_xor((int)bi, m, 32);
            if (ov > bv || (ov == bv && oi < bi)) { bv = ov; bi = oi; }
        }
        if (l15 == 0) {
            unsigned int u = __float_as_uint(bv);
            unsigned int key32 = (u & 0x80000000u) ? ~u : (u | 0x80000000u);
            unsigned long long key =
                ((unsigned long long)key32 << 32) |
                (unsigned long long)(0xFFFFFFFFu - bi);
            int row = m0 + wave * 32 + s * 16 + v + half * 8;
            atomicMax(&best[row], key);
        }
    }
}

// ---------- gather: out[b,:] = W[:, idx_b] ----------
__global__ void gather_k(const float* __restrict__ W,
                         const unsigned long long* __restrict__ best,
                         float* __restrict__ out) {
    int b = blockIdx.x;
    unsigned long long key = best[b];
    unsigned int idx = 0xFFFFFFFFu - (unsigned int)(key & 0xFFFFFFFFull);
    int d = threadIdx.x;
    out[(size_t)b * DIM + d] = W[(size_t)d * KCOLS + idx];
}

extern "C" void kernel_launch(void* const* d_in, const int* in_sizes, int n_in,
                              void* d_out, int out_size, void* d_ws, size_t ws_size,
                              hipStream_t stream) {
    const float* targ = (const float*)d_in[0]; // [B, D]
    const float* W    = (const float*)d_in[1]; // [D, K]
    float*       out  = (float*)d_out;         // [B, D]

    // workspace layout (~38.1 MB)
    unsigned short* Wt = (unsigned short*)d_ws;                         // K*D bf16
    unsigned short* Tb = Wt + (size_t)KCOLS * DIM;                      // B*D bf16
    float*          cninv = (float*)(Tb + (size_t)BATCH * DIM);         // K f32
    unsigned long long* best = (unsigned long long*)(cninv + KCOLS);    // B u64

    prepW_k<<<KCOLS / 256, 256, 0, stream>>>(W, Wt, cninv, best);
    prepT_k<<<(BATCH * DIM / 8) / 256, 256, 0, stream>>>(targ, Tb);

    dim3 grid(KCOLS / (N_TILE * CTILES), BATCH / M_TILE); // 32 x 32
    simargmax_k<<<grid, 256, 0, stream>>>(Tb, Wt, cninv, best);

    gather_k<<<BATCH, DIM, 0, stream>>>(W, best, out);
}
